// transformer_block_61667140435992
// MI455X (gfx1250) — compile-verified, matrix operand order checked by reading, not run on hardware
//
#include <hip/hip_runtime.h>
#include <hip/hip_bf16.h>
#include <math.h>

// Problem constants (match reference)
#define B_   8
#define S_   1024
#define HID_ 2048
#define H_   16
#define D_   128
#define MS_  8192          // B_*S_ rows
#define SCALE_ 0.08838834764831845f  // 1/sqrt(128)

typedef _Float16 v8h  __attribute__((ext_vector_type(8)));
typedef _Float16 v16h __attribute__((ext_vector_type(16)));
typedef float    v8f  __attribute__((ext_vector_type(8)));

union U16 { v16h v; v8h h[2]; };

// Load one 16x32 f16 A/B fragment (ISA 7.12.2 lane layout) from a tile with
// row stride `lda` halves. lanes 0-15: rows 0-15, K 0-7 & 16-23;
// lanes 16-31: rows 0-15, K 8-15 & 24-31. Two 16B chunks -> ds_load_b128.
__device__ __forceinline__ v16h ld_frag(const _Float16* tile, int lda) {
  const int lane = threadIdx.x & 31;
  const _Float16* p = tile + (lane & 15) * lda + (lane >> 4) * 8;
  U16 u;
  u.h[0] = *(const v8h*)(p);
  u.h[1] = *(const v8h*)(p + 16);
  return u.v;
}

__device__ __forceinline__ v8f wmma16(v16h a, v16h b, v8f c) {
  return __builtin_amdgcn_wmma_f32_16x16x32_f16(false, a, false, b, (short)0, c,
                                                false, false);
}

__device__ __forceinline__ v8f zero8() {
  v8f z;
#pragma unroll
  for (int i = 0; i < 8; ++i) z[i] = 0.0f;
  return z;
}

// ---- CDNA5 async DMA: global -> LDS (GLOBAL_LOAD_ASYNC_TO_LDS_B128) -------
// LDS byte address of a __shared__ object: addrspacecast to LDS (AS3,
// 32-bit pointers on device), reinterpret as integer, truncate.
__device__ __forceinline__ unsigned lds_addr_of(const void* p) {
  return (unsigned)(unsigned long long)
         (__attribute__((address_space(3))) const char*)p;
}
__device__ __forceinline__ void async_ld_b128(unsigned lds_off,
                                              const void* gptr) {
  asm volatile("global_load_async_to_lds_b128 %0, %1, off"
               :: "v"(lds_off), "v"(gptr)
               : "memory");
}
__device__ __forceinline__ void wait_async0() {
  asm volatile("s_wait_asynccnt 0x0" ::: "memory");
}
// Wait until at most 4 async loads outstanding: the 4 just-issued next-tile
// loads may still be in flight, but the current tile's DMA has landed
// (ASYNCcnt loads complete in order).
__device__ __forceinline__ void wait_async4() {
  asm volatile("s_wait_asynccnt 0x4" ::: "memory");
}

// ---------------------------------------------------------------------------
// f32 -> f16 conversion (weights)
// ---------------------------------------------------------------------------
__global__ void f32_to_f16(const float* __restrict__ in,
                           _Float16* __restrict__ out, int n) {
  for (int i = blockIdx.x * blockDim.x + threadIdx.x; i < n;
       i += gridDim.x * blockDim.x)
    out[i] = (_Float16)in[i];
}

// ---------------------------------------------------------------------------
// LayerNorm over HID_=2048, one row per block (256 threads, 8 elems/thread).
// Output f16 (feeds WMMA GEMMs).
// ---------------------------------------------------------------------------
__launch_bounds__(256)
__global__ void ln_f16(const float* __restrict__ x, const float* __restrict__ w,
                       _Float16* __restrict__ out) {
  const int row = blockIdx.x;
  const int tid = threadIdx.x;
  const int lane = tid & 31, wv = tid >> 5;
  const float* xr = x + (size_t)row * HID_;

  float v[8], s = 0.f, q = 0.f;
#pragma unroll
  for (int i = 0; i < 8; ++i) {
    v[i] = xr[tid + i * 256];
    s += v[i];
    q += v[i] * v[i];
  }
#pragma unroll
  for (int off = 16; off > 0; off >>= 1) {
    s += __shfl_xor(s, off);
    q += __shfl_xor(q, off);
  }
  __shared__ float ss[8], qq[8];
  if (lane == 0) { ss[wv] = s; qq[wv] = q; }
  __syncthreads();
  if (tid == 0) {
    float S = 0.f, Q = 0.f;
#pragma unroll
    for (int i = 0; i < 8; ++i) { S += ss[i]; Q += qq[i]; }
    ss[0] = S; qq[0] = Q;
  }
  __syncthreads();
  const float mean = ss[0] * (1.0f / HID_);
  const float var  = qq[0] * (1.0f / HID_) - mean * mean;
  const float rstd = rsqrtf(var + 1e-5f);
#pragma unroll
  for (int i = 0; i < 8; ++i) {
    const int c = tid + i * 256;
    out[(size_t)row * HID_ + c] = (_Float16)((v[i] - mean) * rstd * w[c]);
  }
}

// ---------------------------------------------------------------------------
// WMMA GEMM: C[M,N] = A[M,K] * Bw[N,K]^T   (both K-contiguous, f16)
// Block tile 128x128, 8 waves (each 32x64 via 2x4 WMMA frags), BLOCK_K=32.
// Double-buffered GLOBAL_LOAD_ASYNC_TO_LDS_B128 staging: tile t+1's DMA
// overlaps tile t's WMMA work (s_wait_asynccnt 4 threshold).
// MODE 0: outf = C          (f32)
// MODE 1: outf = res + C    (f32, residual add)
// MODE 2: outh = gelu(C)    (f16, exact gelu)
// ---------------------------------------------------------------------------
template <int MODE>
__launch_bounds__(256)
__global__ void gemm_f16_wmma(const _Float16* __restrict__ A,
                              const _Float16* __restrict__ Bw,
                              float* __restrict__ outf,
                              _Float16* __restrict__ outh,
                              const float* __restrict__ res,
                              int M, int N, int K) {
  __shared__ _Float16 As[2][128 * 40];   // pad to 40 halves, 16B-aligned rows
  __shared__ _Float16 Bs[2][128 * 40];

  const int tid = threadIdx.x;
  const int lane = tid & 31, wv = tid >> 5;
  const int l16 = lane & 15, hf = lane >> 4;
  const int wm = wv >> 1, wn = wv & 1;            // 4x2 wave grid
  const int bm = blockIdx.y * 128, bn = blockIdx.x * 128;
  const int arow = tid >> 1, acol = (tid & 1) * 16;

  v8f acc[2][4];
#pragma unroll
  for (int i = 0; i < 2; ++i)
#pragma unroll
    for (int j = 0; j < 4; ++j) acc[i][j] = zero8();

  const _Float16* ap = A  + (size_t)(bm + arow) * K + acol;
  const _Float16* bp = Bw + (size_t)(bn + arow) * K + acol;
  const unsigned lA[2] = { lds_addr_of(As[0] + arow * 40 + acol),
                           lds_addr_of(As[1] + arow * 40 + acol) };
  const unsigned lB[2] = { lds_addr_of(Bs[0] + arow * 40 + acol),
                           lds_addr_of(Bs[1] + arow * 40 + acol) };

  const int nt = K >> 5;               // tiles of BLOCK_K=32
  // Prologue: DMA tile 0 into buffer 0.
  async_ld_b128(lA[0],      ap);
  async_ld_b128(lA[0] + 16, ap + 8);
  async_ld_b128(lB[0],      bp);
  async_ld_b128(lB[0] + 16, bp + 8);

  for (int t = 0; t < nt; ++t) {
    const int cur = t & 1;
    if (t + 1 < nt) {
      const int nxt = cur ^ 1;
      const int k1 = (t + 1) << 5;
      __builtin_prefetch(ap + k1 + 128, 0, 1);  // global_prefetch_b8 ahead
      __builtin_prefetch(bp + k1 + 128, 0, 1);
      // Issue tile t+1 into the other buffer (its readers finished at the
      // barrier that ended iteration t-1), then wait only for tile t's DMA.
      async_ld_b128(lA[nxt],      ap + k1);
      async_ld_b128(lA[nxt] + 16, ap + k1 + 8);
      async_ld_b128(lB[nxt],      bp + k1);
      async_ld_b128(lB[nxt] + 16, bp + k1 + 8);
      wait_async4();
    } else {
      wait_async0();
    }
    __syncthreads();                   // all threads' tile-t DMA landed

    v16h af[2], bf[4];
#pragma unroll
    for (int i = 0; i < 2; ++i)
      af[i] = ld_frag(As[cur] + (wm * 32 + i * 16) * 40, 40);
#pragma unroll
    for (int j = 0; j < 4; ++j)
      bf[j] = ld_frag(Bs[cur] + (wn * 64 + j * 16) * 40, 40);
#pragma unroll
    for (int i = 0; i < 2; ++i)
#pragma unroll
      for (int j = 0; j < 4; ++j) acc[i][j] = wmma16(af[i], bf[j], acc[i][j]);

    __syncthreads();                   // reads done before buffer reuse
  }

#pragma unroll
  for (int i = 0; i < 2; ++i)
#pragma unroll
    for (int j = 0; j < 4; ++j)
#pragma unroll
      for (int jj = 0; jj < 8; ++jj) {
        const int r = bm + wm * 32 + i * 16 + jj + hf * 8;
        const int c = bn + wn * 64 + j * 16 + l16;
        const size_t o = (size_t)r * N + c;
        const float vv = acc[i][j][jj];
        if (MODE == 0)      outf[o] = vv;
        else if (MODE == 1) outf[o] = res[o] + vv;
        else outh[o] = (_Float16)(0.5f * vv * (1.0f + erff(vv * 0.70710678118f)));
      }
}

// ---------------------------------------------------------------------------
// xPos rotary for K and V (V divides by zeta^seq), plus layout conversion:
//   qT,kT : [B,H,S,D] f16        vT : [B,H,D,S] f16 (pre-transposed for P*V)
// One thread per (b,s,h,pair). 8*1024*16*64 threads.
// ---------------------------------------------------------------------------
__global__ void rope_xpos(const float* __restrict__ q,
                          const float* __restrict__ k,
                          const float* __restrict__ v,
                          const int* __restrict__ idxp,
                          _Float16* __restrict__ qT,
                          _Float16* __restrict__ kT,
                          _Float16* __restrict__ vT) {
  const int gid = blockIdx.x * blockDim.x + threadIdx.x;
  const int j = gid & 63;
  const int h = (gid >> 6) & 15;
  const int s = (gid >> 10) & 1023;
  const int b = gid >> 20;
  const int index = idxp[0];

  const size_t src = ((size_t)(b * S_ + s)) * HID_ + h * D_ + 2 * j;
  const float q0 = q[src], q1 = q[src + 1];
  const float k0 = k[src], k1 = k[src + 1];
  const float v0 = v[src], v1 = v[src + 1];

  const float dr   = 2.0f + 2.0f * (float)j;                 // drange element
  const float seq  = ((float)(index + s) - 512.0f) * (1.0f / 512.0f);
  const float th   = __powf(1.0f / 10000.0f, dr * (1.0f / 128.0f));
  const float c    = __cosf(seq * th);
  const float sn   = __sinf(seq * th);
  const float zeta = (dr * (1.0f / 64.0f) + 51.2f) * (1.0f / 52.2f);
  const float t    = __powf(zeta, seq);
  const float it   = 1.0f / t;

  const float kr0 = (k0 * c - k1 * sn) * t;
  const float kr1 = (k1 * c + k0 * sn) * t;
  const float vr0 = (v0 * c - v1 * sn) * it;
  const float vr1 = (v1 * c + v0 * sn) * it;

  const size_t bh = (size_t)(b * H_ + h);
  const size_t dq = (bh * S_ + s) * D_ + 2 * j;
  qT[dq]     = (_Float16)q0;  qT[dq + 1] = (_Float16)q1;
  kT[dq]     = (_Float16)kr0; kT[dq + 1] = (_Float16)kr1;
  const size_t dv = (bh * D_ + 2 * j) * S_ + s;
  vT[dv]      = (_Float16)vr0;
  vT[dv + S_] = (_Float16)vr1;
}

// ---------------------------------------------------------------------------
// Flash attention, one block per (q-block of 64, h, b). 4 waves / 128 thr.
// Q fragments live in registers; K tile & V^T tile staged in LDS via async
// DMA; S tile via WMMA -> LDS f32; online softmax; P*V via WMMA into
// register accumulators.
// ---------------------------------------------------------------------------
__launch_bounds__(128)
__global__ void attn_flash(const _Float16* __restrict__ qT,
                           const _Float16* __restrict__ kT,
                           const _Float16* __restrict__ vT,
                           _Float16* __restrict__ ctx) {
  __shared__ _Float16 Ks[64 * 136];   // [kRow][d], pad 136
  __shared__ _Float16 Vs[128 * 72];   // [d][kRow], pad 72
  __shared__ float    Ss[64 * 68];    // scores f32
  __shared__ _Float16 Ps[64 * 72];    // probabilities f16
  __shared__ float m_s[64], l_s[64], r_s[64];

  const int tid = threadIdx.x;
  const int lane = tid & 31, wv = tid >> 5;
  const int l16 = lane & 15, hf = lane >> 4;
  const int qb = blockIdx.x, h = blockIdx.y, b = blockIdx.z;
  const size_t bh = (size_t)b * H_ + h;

  // Q fragments: this wave owns rows qb*64 + wv*16 .. +15, K split in 4x32.
  v16h qf[4];
  {
    const _Float16* qbase = qT + (bh * S_ + qb * 64 + wv * 16) * D_;
#pragma unroll
    for (int kst = 0; kst < 4; ++kst) qf[kst] = ld_frag(qbase + kst * 32, D_);
  }
  if (tid < 64) { m_s[tid] = -INFINITY; l_s[tid] = 0.0f; }

  v8f o[8];
#pragma unroll
  for (int f = 0; f < 8; ++f) o[f] = zero8();

  for (int kb = 0; kb <= qb; ++kb) {
    __syncthreads();   // prior-iteration LDS consumers done
    // Stage K tile [64][128] and V^T tile [128][64] via async DMA to LDS;
    // prefetch the next K/V tiles into L2 while we work on this one.
#pragma unroll
    for (int i = 0; i < 8; ++i) {
      const int cidx = (i * 128 + tid) * 8;
      const int row = cidx >> 7, col = cidx & 127;
      const _Float16* kp = kT + (bh * S_ + kb * 64 + row) * D_ + col;
      async_ld_b128(lds_addr_of(Ks + row * 136 + col), kp);
      __builtin_prefetch(kp + 64 * D_, 0, 1);
      const int d = cidx >> 6, kk = cidx & 63;
      const _Float16* vp = vT + (bh * D_ + d) * S_ + kb * 64 + kk;
      async_ld_b128(lds_addr_of(Vs + d * 72 + kk), vp);
      __builtin_prefetch(vp + 64, 0, 1);
    }
    wait_async0();
    __syncthreads();

    // S tile: wave rows x 64 cols
#pragma unroll
    for (int cf = 0; cf < 4; ++cf) {
      v8f s = zero8();
#pragma unroll
      for (int kst = 0; kst < 4; ++kst)
        s = wmma16(qf[kst], ld_frag(Ks + cf * 16 * 136 + kst * 32, 136), s);
#pragma unroll
      for (int jj = 0; jj < 8; ++jj)
        Ss[(wv * 16 + jj + hf * 8) * 68 + cf * 16 + l16] = s[jj];
    }
    __syncthreads();

    // Online softmax, one thread per row
    if (tid < 64) {
      const int r = tid, qg = qb * 64 + r;
      const float mold = m_s[r];
      float mx = mold;
      for (int c = 0; c < 64; ++c) {
        const int kg = kb * 64 + c;
        const float sv = (kg <= qg) ? Ss[r * 68 + c] * SCALE_ : -INFINITY;
        mx = fmaxf(mx, sv);
      }
      const float resc = __expf(mold - mx);
      float sum = 0.0f;
      for (int c = 0; c < 64; ++c) {
        const int kg = kb * 64 + c;
        const float e = (kg <= qg) ? __expf(Ss[r * 68 + c] * SCALE_ - mx) : 0.0f;
        sum += e;
        Ps[r * 72 + c] = (_Float16)e;
      }
      l_s[r] = l_s[r] * resc + sum;
      m_s[r] = mx;
      r_s[r] = resc;
    }
    __syncthreads();

    // Rescale accumulators and add P*V
    float rr[8];
#pragma unroll
    for (int jj = 0; jj < 8; ++jj) rr[jj] = r_s[wv * 16 + jj + hf * 8];
    v16h pf[2];
    pf[0] = ld_frag(Ps + wv * 16 * 72, 72);
    pf[1] = ld_frag(Ps + wv * 16 * 72 + 32, 72);
#pragma unroll
    for (int df = 0; df < 8; ++df) {
#pragma unroll
      for (int jj = 0; jj < 8; ++jj) o[df][jj] *= rr[jj];
      o[df] = wmma16(pf[0], ld_frag(Vs + df * 16 * 72, 72), o[df]);
      o[df] = wmma16(pf[1], ld_frag(Vs + df * 16 * 72 + 32, 72), o[df]);
    }
  }

  float li[8];
#pragma unroll
  for (int jj = 0; jj < 8; ++jj) li[jj] = 1.0f / l_s[wv * 16 + jj + hf * 8];
#pragma unroll
  for (int df = 0; df < 8; ++df)
#pragma unroll
    for (int jj = 0; jj < 8; ++jj) {
      const int srow = qb * 64 + wv * 16 + jj + hf * 8;
      const int col = h * D_ + df * 16 + l16;
      ctx[((size_t)b * S_ + srow) * HID_ + col] = (_Float16)(o[df][jj] * li[jj]);
    }
}

// ---------------------------------------------------------------------------
// Host orchestration
// ---------------------------------------------------------------------------
extern "C" void kernel_launch(void* const* d_in, const int* in_sizes, int n_in,
                              void* d_out, int out_size, void* d_ws,
                              size_t ws_size, hipStream_t stream) {
  (void)in_sizes; (void)n_in; (void)out_size; (void)ws_size;

  const float* acts0  = (const float*)d_in[0];
  const float* cachek = (const float*)d_in[1];
  const float* cachev = (const float*)d_in[2];
  const int*   idxp   = (const int*)d_in[3];
  const float* ln1w   = (const float*)d_in[4];
  const float* ln2w   = (const float*)d_in[5];
  const float* qw     = (const float*)d_in[6];
  const float* kw     = (const float*)d_in[7];
  const float* vw     = (const float*)d_in[8];
  const float* ow     = (const float*)d_in[9];
  const float* w1     = (const float*)d_in[10];
  const float* w2     = (const float*)d_in[11];

  char* ws = (char*)d_ws;
  size_t off = 0;
  auto alloc = [&](size_t bytes) -> void* {
    void* p = ws + off;
    off += (bytes + 255) & ~(size_t)255;
    return p;
  };

  const size_t HH  = (size_t)HID_ * HID_;        // 4.19M
  const size_t MH  = (size_t)MS_ * HID_;         // 16.78M (rows x 2048)
  _Float16* xn1_h = (_Float16*)alloc(MH * 2);
  _Float16* qw_h  = (_Float16*)alloc(HH * 2);
  _Float16* kw_h  = (_Float16*)alloc(HH * 2);
  _Float16* vw_h  = (_Float16*)alloc(HH * 2);
  _Float16* ow_h  = (_Float16*)alloc(HH * 2);
  _Float16* w1_h  = (_Float16*)alloc(4 * HH * 2);
  _Float16* w2_h  = (_Float16*)alloc(4 * HH * 2);
  // big region: q/k/v f32 (3 x 64MB); recycled after rope as h_h (128MB) +
  // acts1 (64MB) with no live overlap.
  char* big = (char*)alloc(3 * MH * 4);
  float*    q_f   = (float*)big;
  float*    k_f   = (float*)(big + MH * 4);
  float*    v_f   = (float*)(big + 2 * MH * 4);
  _Float16* h_h   = (_Float16*)big;              // [8192,8192] f16 = 2*MH*4 B
  float*    acts1 = (float*)(big + 2 * MH * 4);  // reuses v_f region
  _Float16* qT    = (_Float16*)alloc(MH * 2);
  _Float16* kTb   = (_Float16*)alloc(MH * 2);
  _Float16* vTb   = (_Float16*)alloc(MH * 2);
  _Float16* ctx_h = (_Float16*)alloc(MH * 2);
  _Float16* xn2_h = (_Float16*)alloc(MH * 2);

  // Convert weights f32 -> f16
  f32_to_f16<<<2048, 256, 0, stream>>>(qw, qw_h, (int)HH);
  f32_to_f16<<<2048, 256, 0, stream>>>(kw, kw_h, (int)HH);
  f32_to_f16<<<2048, 256, 0, stream>>>(vw, vw_h, (int)HH);
  f32_to_f16<<<2048, 256, 0, stream>>>(ow, ow_h, (int)HH);
  f32_to_f16<<<4096, 256, 0, stream>>>(w1, w1_h, (int)(4 * HH));
  f32_to_f16<<<4096, 256, 0, stream>>>(w2, w2_h, (int)(4 * HH));

  // LN1
  ln_f16<<<MS_, 256, 0, stream>>>(acts0, ln1w, xn1_h);

  // QKV projections (f32 out)
  gemm_f16_wmma<0><<<dim3(16, 64), 256, 0, stream>>>(xn1_h, qw_h, q_f, nullptr,
                                                     nullptr, MS_, HID_, HID_);
  gemm_f16_wmma<0><<<dim3(16, 64), 256, 0, stream>>>(xn1_h, kw_h, k_f, nullptr,
                                                     nullptr, MS_, HID_, HID_);
  gemm_f16_wmma<0><<<dim3(16, 64), 256, 0, stream>>>(xn1_h, vw_h, v_f, nullptr,
                                                     nullptr, MS_, HID_, HID_);

  // xPos rotary + layout conversion to f16
  rope_xpos<<<32768, 256, 0, stream>>>(q_f, k_f, v_f, idxp, qT, kTb, vTb);

  // Flash attention
  attn_flash<<<dim3(S_ / 64, H_, B_), 128, 0, stream>>>(qT, kTb, vTb, ctx_h);

  // Output projection + residual -> acts1 (f32)
  gemm_f16_wmma<1><<<dim3(16, 64), 256, 0, stream>>>(ctx_h, ow_h, acts1,
                                                     nullptr, acts0, MS_, HID_,
                                                     HID_);

  // LN2 -> f16
  ln_f16<<<MS_, 256, 0, stream>>>(acts1, ln2w, xn2_h);

  // MLP up + exact GELU -> h_h (f16)
  gemm_f16_wmma<2><<<dim3(64, 64), 256, 0, stream>>>(xn2_h, w1_h, nullptr, h_h,
                                                     nullptr, MS_, 4 * HID_,
                                                     HID_);

  // MLP down + residual -> final acts directly into d_out (f32)
  gemm_f16_wmma<1><<<dim3(16, 64), 256, 0, stream>>>(h_h, w2_h, (float*)d_out,
                                                     nullptr, acts1, MS_, HID_,
                                                     4 * HID_);

  // Pass-through caches into the concatenated output
  const size_t actsB  = (size_t)MS_ * HID_ * 4;
  const size_t cacheB = (size_t)B_ * S_ * H_ * D_ * 4;
  (void)hipMemcpyAsync((char*)d_out + actsB, cachek, cacheB,
                       hipMemcpyDeviceToDevice, stream);
  (void)hipMemcpyAsync((char*)d_out + actsB + cacheB, cachev, cacheB,
                       hipMemcpyDeviceToDevice, stream);
}
